// unit_gcn_76922864272134
// MI455X (gfx1250) — matrix operand order repaired and checked by reading, not compiled, not run on hardware
//
#include <hip/hip_runtime.h>
#include <hip/hip_bf16.h>

typedef __attribute__((ext_vector_type(16))) _Float16 v16h;
typedef __attribute__((ext_vector_type(8)))  _Float16 v8h;
typedef __attribute__((ext_vector_type(8)))  float    v8f;

// ---- problem constants ----
#define NN 128
#define CIN 64
#define COUT 64
#define TT 128
#define VV 25
#define KK 3
#define GG 8
#define RR 8
#define TV 3200              // T*V
#define D3 192               // K*COUT
#define MTOT 409600ULL       // N*T*V

// Build a v16h fragment from two contiguous 16-byte runs.
__device__ __forceinline__ v16h frag2(v8h a, v8h b) {
    return __builtin_shufflevector(a, b, 0, 1, 2, 3, 4, 5, 6, 7, 8, 9, 10, 11, 12, 13, 14, 15);
}
// A-operand fragment: lane needs K = {base..base+15} contiguous (16B-aligned base).
__device__ __forceinline__ v16h loadA(const _Float16* base) {
    return frag2(*(const v8h*)base, *(const v8h*)(base + 8));
}
// B-operand fragment from N-major storage: 16 contiguous K values at base.
__device__ __forceinline__ v16h loadB(const _Float16* base) {
    return frag2(*(const v8h*)base, *(const v8h*)(base + 8));
}

// ---------------- small prep kernels ----------------

__global__ void k_zero(float* p, int n) {
    int i = blockIdx.x * 256 + threadIdx.x;
    if (i < n) p[i] = 0.f;
}

// zero the v=25..31 pad slots of [rows][32] f16 tensors
__global__ void k_padzero(_Float16* p, long rows) {
    long i = (long)blockIdx.x * 256 + threadIdx.x;
    if (i >= rows) return;
    _Float16* r = p + i * 32;
    r[25] = (_Float16)0.f;
    *(unsigned int*)(&r[26]) = 0u;
    *(unsigned int*)(&r[28]) = 0u;
    *(unsigned int*)(&r[30]) = 0u;
}

// normA[k,g,v,w] = Dec / (colsum_v + 1e-3); S[k,g,w] = colsum(normA)
__global__ void k_norm_adj(const float* __restrict__ Dec, float* normA, float* S) {
    int kg = blockIdx.x;                // 24 = K*G
    int w = threadIdx.x;
    if (w >= VV) return;
    const float* Ag = Dec + (size_t)kg * (VV * VV);
    float d = 0.f;
    for (int v = 0; v < VV; ++v) d += Ag[v * VV + w];
    float inv = 1.f / (d + 0.001f);
    for (int v = 0; v < VV; ++v) normA[(size_t)kg * (VV * VV) + v * VV + w] = Ag[v * VV + w] * inv;
    S[kg * VV + w] = d * inv;
}

// xm[n,c,v] = mean_t x0[n,c,t,v]
__global__ void k_mean(const float* __restrict__ x0, float* xm) {
    int i = blockIdx.x * 256 + threadIdx.x;
    if (i >= NN * CIN * VV) return;
    int v = i % VV;
    int nc = i / VV;
    const float* p = x0 + (size_t)nc * TV + v;
    float s = 0.f;
    for (int t = 0; t < TT; ++t) s += p[t * VV];
    xm[i] = s * (1.f / (float)TT);
}

// attention, stored TRANSPOSED (N-major): attT[k,n,o, w, u], 32x32 zero-padded f16
__global__ void k_att(const float* __restrict__ xm,
                      const float* __restrict__ w1, const float* __restrict__ b1,
                      const float* __restrict__ w2, const float* __restrict__ b2,
                      const float* __restrict__ w4, const float* __restrict__ b4,
                      const float* __restrict__ alpha, const float* __restrict__ A,
                      _Float16* __restrict__ attT) {
    int k = blockIdx.x % KK;
    int n = blockIdx.x / KK;
    __shared__ float xs[CIN][VV];
    __shared__ float x1s[RR][VV], x2s[RR][VV];
    __shared__ float diffs[RR][VV][VV];
    int tid = threadIdx.x;
    for (int i = tid; i < CIN * VV; i += 256) xs[i / VV][i % VV] = xm[(size_t)n * (CIN * VV) + i];
    __syncthreads();
    for (int idx = tid; idx < 2 * RR * VV; idx += 256) {
        int which = idx / (RR * VV);
        int rem = idx % (RR * VV);
        int r = rem / VV, u = rem % VV;
        const float* w = which ? w2 : w1;
        const float* bb = which ? b2 : b1;
        float s = bb[k * RR + r];
        for (int c = 0; c < CIN; ++c) s += w[(k * RR + r) * CIN + c] * xs[c][u];
        (which ? x2s : x1s)[r][u] = s;
    }
    __syncthreads();
    for (int idx = tid; idx < RR * VV * VV; idx += 256) {
        int r = idx / (VV * VV);
        int u = (idx / VV) % VV;
        int w = idx % VV;
        diffs[r][u][w] = tanhf(x1s[r][u] - x2s[r][w]);
    }
    __syncthreads();
    float al = alpha[0];
    for (int i = tid; i < COUT * 32 * 32; i += 256) {
        int o = i >> 10, u = (i >> 5) & 31, w = i & 31;
        float val = 0.f;
        if (u < VV && w < VV) {
            float s = b4[k * COUT + o];
            for (int r = 0; r < RR; ++r) s += w4[(k * COUT + o) * RR + r] * diffs[r][u][w];
            val = al * s + A[(k * VV + u) * VV + w];
        }
        // transposed store: [w][u]
        attT[((size_t)(k * NN + n) * COUT + o) * 1024 + w * 32 + u] = (_Float16)val;
    }
}

// ---------------- big WMMA GEMMs ----------------

// XlinP[n,d,t,0..31] = sum_c x0[n,c,t,v]*Lw[c,d] + Lb[d]; BN0 stats via LDS float atomics.
__global__ void k_lin_gemm(const float* __restrict__ x0, const float* __restrict__ Lw,
                           const float* __restrict__ Lb, _Float16* __restrict__ XlinP,
                           float* sum0, float* sq0) {
    int sb = blockIdx.x, dt = blockIdx.y, n = blockIdx.z;
    int dbase = dt * 16, s0 = sb * 128;
    __shared__ _Float16 At[16][64];
    __shared__ float ssum[16], ssq[16];
    int tid = threadIdx.x;
    if (tid < 16) { ssum[tid] = 0.f; ssq[tid] = 0.f; }
    for (int i = tid; i < 16 * 64; i += 256) {
        int r = i >> 6, c = i & 63;
        At[r][c] = (_Float16)Lw[c * D3 + dbase + r];
    }
    __syncthreads();
    int wv = tid >> 5, lane = tid & 31, m = lane & 15;
    bool hi = lane >= 16;
    int scol = s0 + wv * 16 + m;
    v8f acc;
    for (int r = 0; r < 8; ++r) acc[r] = Lb[dbase + (hi ? 8 + r : r)];
    const float* bsrc = x0 + (size_t)n * CIN * TV + scol;
    for (int kb = 0; kb < 64; kb += 32) {
        v16h a = loadA(&At[m][kb + (hi ? 16 : 0)]);
        v16h b;
        int cb = kb + (hi ? 16 : 0);
#pragma unroll
        for (int i = 0; i < 16; ++i) b[i] = (_Float16)bsrc[(size_t)(cb + i) * TV];
        acc = __builtin_amdgcn_wmma_f32_16x16x32_f16(false, a, false, b, (short)0, acc, false, false);
    }
    int t = scol / VV, v = scol % VV;
    for (int r = 0; r < 8; ++r) {
        int row = hi ? 8 + r : r;
        float val = acc[r];
        XlinP[(((size_t)n * D3 + dbase + row) * TT + t) * 32 + v] = (_Float16)val;
        atomicAdd(&ssum[row], val);
        atomicAdd(&ssq[row], val * val);
    }
    __syncthreads();
    if (tid < 16) {
        atomicAdd(&sum0[dbase + tid], ssum[tid]);
        atomicAdd(&sq0[dbase + tid], ssq[tid]);
    }
}

// finalize BN affine: a=gamma*rsqrt(var+eps), b=beta-mu*a
__global__ void k_finalize_bn(const float* sum, const float* sq, const float* gamma,
                              const float* beta, float* a, float* b, int C, float invM) {
    int d = blockIdx.x * blockDim.x + threadIdx.x;
    if (d >= C) return;
    float mu = sum[d] * invM;
    float var = sq[d] * invM - mu * mu;
    float s = gamma[d] * rsqrtf(var + 1e-5f);
    a[d] = s;
    b[d] = beta[d] - mu * s;
}

// BscT[k,c, ww, vv] = a0[k*64+c] * normA[k,c%8,vv,ww]  (transposed, zero padded 32x32)
__global__ void k_bscale(const float* __restrict__ normA, const float* __restrict__ a0,
                         _Float16* __restrict__ BscT) {
    int kc = blockIdx.x;                 // 192
    int k = kc >> 6, c = kc & 63, g = c & 7;
    float a = a0[kc];
    for (int i = threadIdx.x; i < 1024; i += blockDim.x) {
        int ww = i >> 5, vv = i & 31;
        float val = (vv < VV && ww < VV) ? a * normA[(size_t)(k * GG + g) * (VV * VV) + vv * VV + ww] : 0.f;
        BscT[(size_t)kc * 1024 + i] = (_Float16)val;
    }
}

// off[c,w] = sum_k b0[k*64+c] * S[k, c%8, w]
__global__ void k_off(const float* __restrict__ S, const float* __restrict__ b0, float* off) {
    int c = blockIdx.x;
    int ww = threadIdx.x;                // 32
    float v = 0.f;
    if (ww < VV)
        for (int k = 0; k < KK; ++k) v += b0[k * COUT + c] * S[(k * GG + (c & 7)) * VV + ww];
    off[c * 32 + ww] = v;
}

// x[n,c,t,w] = sum_k sum_v XlinP[n,k*64+c,t,v] * BscT[k,c,w,v] + off[c,w]  -> catT[n][s][c]
__global__ void k_adj(const _Float16* __restrict__ XlinP, const _Float16* __restrict__ BscT,
                      const float* __restrict__ off, _Float16* __restrict__ catT) {
    int tt = blockIdx.x, c = blockIdx.y, n = blockIdx.z;
    int t0 = tt * 16;
    int lane = threadIdx.x & 31, wv = threadIdx.x >> 5;
    int m = lane & 15;
    bool hi = lane >= 16;
    int wcol = wv * 16 + m;
    float offv = (wcol < VV) ? off[c * 32 + wcol] : 0.f;
    v8f acc;
    for (int r = 0; r < 8; ++r) acc[r] = offv;
    for (int k = 0; k < KK; ++k) {
        const _Float16* arow = XlinP + ((size_t)n * D3 + k * COUT + c) * TT * 32;
        const _Float16* brow = BscT + ((size_t)(k * COUT + c) * 32 + wcol) * 32;
        v16h a = loadA(arow + (t0 + m) * 32 + (hi ? 16 : 0));
        v16h b = loadB(brow + (hi ? 16 : 0));
        acc = __builtin_amdgcn_wmma_f32_16x16x32_f16(false, a, false, b, (short)0, acc, false, false);
    }
    if (wcol < VV)
        for (int r = 0; r < 8; ++r) {
            int t = t0 + (hi ? 8 + r : r);
            catT[((size_t)n * TV + t * VV + wcol) * 128 + c] = (_Float16)acc[r];
        }
}

// x3P[k,n,o,t,0..31] = sum_c x0[n,c,t,v]*w3[k,o,c] + b3[k,o]
__global__ void k_x3_gemm(const float* __restrict__ x0, const float* __restrict__ w3,
                          const float* __restrict__ b3, _Float16* __restrict__ x3P) {
    int sb = blockIdx.x, ko = blockIdx.y, n = blockIdx.z;
    int k = ko >> 2, ot = ko & 3;
    int obase = ot * 16, s0 = sb * 128, tid = threadIdx.x;
    __shared__ _Float16 At[16][64];
    for (int i = tid; i < 16 * 64; i += 256) {
        int r = i >> 6, c = i & 63;
        At[r][c] = (_Float16)w3[(k * COUT + obase + r) * CIN + c];
    }
    __syncthreads();
    int wv = tid >> 5, lane = tid & 31, m = lane & 15;
    bool hi = lane >= 16;
    int scol = s0 + wv * 16 + m;
    v8f acc;
    for (int r = 0; r < 8; ++r) acc[r] = b3[k * COUT + obase + (hi ? 8 + r : r)];
    const float* bsrc = x0 + (size_t)n * CIN * TV + scol;
    for (int kb = 0; kb < 64; kb += 32) {
        v16h a = loadA(&At[m][kb + (hi ? 16 : 0)]);
        v16h b;
        int cb = kb + (hi ? 16 : 0);
#pragma unroll
        for (int i = 0; i < 16; ++i) b[i] = (_Float16)bsrc[(size_t)(cb + i) * TV];
        acc = __builtin_amdgcn_wmma_f32_16x16x32_f16(false, a, false, b, (short)0, acc, false, false);
    }
    int t = scol / VV, v = scol % VV;
    for (int r = 0; r < 8; ++r) {
        int o = obase + (hi ? 8 + r : r);
        x3P[(((size_t)(k * NN + n) * COUT + o) * TT + t) * 32 + v] = (_Float16)acc[r];
    }
}

// y[n,c,t,v] = sum_k sum_u x3P[k,n,c,t,u] * attT[k,n,c,v,u] -> catT[n][s][64+c]
__global__ void k_y_gemm(const _Float16* __restrict__ x3P, const _Float16* __restrict__ attT,
                         _Float16* __restrict__ catT) {
    int tt = blockIdx.x, c = blockIdx.y, n = blockIdx.z;
    int t0 = tt * 16;
    int lane = threadIdx.x & 31, wv = threadIdx.x >> 5;
    int m = lane & 15;
    bool hi = lane >= 16;
    int wcol = wv * 16 + m;
    v8f acc;
    for (int r = 0; r < 8; ++r) acc[r] = 0.f;
    for (int k = 0; k < KK; ++k) {
        const _Float16* arow = x3P + ((size_t)(k * NN + n) * COUT + c) * TT * 32;
        const _Float16* brow = attT + ((size_t)(k * NN + n) * COUT + c) * 1024 + wcol * 32;
        v16h a = loadA(arow + (t0 + m) * 32 + (hi ? 16 : 0));
        v16h b = loadB(brow + (hi ? 16 : 0));
        acc = __builtin_amdgcn_wmma_f32_16x16x32_f16(false, a, false, b, (short)0, acc, false, false);
    }
    if (wcol < VV)
        for (int r = 0; r < 8; ++r) {
            int t = t0 + (hi ? 8 + r : r);
            catT[((size_t)n * TV + t * VV + wcol) * 128 + 64 + c] = (_Float16)acc[r];
        }
}

// out_pre[n,o,s] = sum_c ens_w[o,c]*catT[n,s,c] + ens_b[o]; final BN stats.
// B panel staged to LDS with CDNA5 async global->LDS copies (ASYNCcnt).
__global__ void k_ens_gemm(const _Float16* __restrict__ catT, const float* __restrict__ ew,
                           const float* __restrict__ eb, _Float16* __restrict__ outp,
                           float* sum1, float* sq1) {
    int sb = blockIdx.x, ot = blockIdx.y, n = blockIdx.z;
    int obase = ot * 16, s0 = sb * 128, tid = threadIdx.x;
    __shared__ _Float16 At[16][128];
    __shared__ _Float16 Bt[128][128];
    __shared__ float ssum[16], ssq[16];
    if (tid < 16) { ssum[tid] = 0.f; ssq[tid] = 0.f; }
    for (int i = tid; i < 16 * 128; i += 256) {
        int r = i >> 7, c = i & 127;
        At[r][c] = (_Float16)ew[(obase + r) * 128 + c];
    }
    // async stage: 128 rows x 256B, 16B chunks
    const _Float16* gsrc = catT + ((size_t)n * TV + s0) * 128;
    for (int q = tid; q < 2048; q += 256) {
        int j = q >> 4, ch = q & 15;
        unsigned lds = (unsigned)(size_t)(&Bt[j][ch * 8]);
        const _Float16* g = gsrc + (size_t)j * 128 + ch * 8;
        asm volatile("global_load_async_to_lds_b128 %0, %1, off" :: "v"(lds), "v"(g) : "memory");
    }
    asm volatile("s_wait_asynccnt 0" ::: "memory");
    __syncthreads();
    int wv = tid >> 5, lane = tid & 31, m = lane & 15;
    bool hi = lane >= 16;
    v8f acc;
    for (int r = 0; r < 8; ++r) acc[r] = eb[obase + (hi ? 8 + r : r)];
    for (int kb = 0; kb < 128; kb += 32) {
        v16h a = loadA(&At[m][kb + (hi ? 16 : 0)]);
        v16h b = loadB(&Bt[wv * 16 + m][kb + (hi ? 16 : 0)]);
        acc = __builtin_amdgcn_wmma_f32_16x16x32_f16(false, a, false, b, (short)0, acc, false, false);
    }
    int scol = s0 + wv * 16 + m;
    for (int r = 0; r < 8; ++r) {
        int row = hi ? 8 + r : r;
        float v = acc[r];
        outp[((size_t)n * COUT + obase + row) * TV + scol] = (_Float16)v;
        atomicAdd(&ssum[row], v);
        atomicAdd(&ssq[row], v * v);
    }
    __syncthreads();
    if (tid < 16) {
        atomicAdd(&sum1[obase + tid], ssum[tid]);
        atomicAdd(&sq1[obase + tid], ssq[tid]);
    }
}

// out = relu(bn(out_pre) + x0)
__global__ void k_final(const _Float16* __restrict__ outp, const float* __restrict__ x0,
                        const float* __restrict__ a2, const float* __restrict__ b2,
                        float* __restrict__ out) {
    size_t i = (size_t)blockIdx.x * 256 + threadIdx.x;
    if (i >= (size_t)NN * COUT * TV) return;
    int o = (int)((i / TV) % COUT);
    float v = a2[o] * (float)outp[i] + b2[o] + x0[i];
    out[i] = v > 0.f ? v : 0.f;
}

// ---------------- launch ----------------

extern "C" void kernel_launch(void* const* d_in, const int* in_sizes, int n_in,
                              void* d_out, int out_size, void* d_ws, size_t ws_size,
                              hipStream_t stream) {
    const float* x0    = (const float*)d_in[0];
    const float* Dec   = (const float*)d_in[1];
    const float* Aadj  = (const float*)d_in[2];
    const float* alpha = (const float*)d_in[3];
    const float* w1 = (const float*)d_in[4];
    const float* b1 = (const float*)d_in[5];
    const float* w2 = (const float*)d_in[6];
    const float* b2 = (const float*)d_in[7];
    const float* w3 = (const float*)d_in[8];
    const float* b3 = (const float*)d_in[9];
    const float* w4 = (const float*)d_in[10];
    const float* b4 = (const float*)d_in[11];
    const float* Lw = (const float*)d_in[12];
    const float* Lb = (const float*)d_in[13];
    const float* g0 = (const float*)d_in[14];
    const float* be0 = (const float*)d_in[15];
    const float* ew = (const float*)d_in[16];
    const float* eb = (const float*)d_in[17];
    const float* g1 = (const float*)d_in[18];
    const float* be1 = (const float*)d_in[19];
    float* out = (float*)d_out;

    char* ws = (char*)d_ws;
    size_t cur = 0;
    auto take = [&](size_t bytes) {
        size_t o = cur;
        cur = (cur + bytes + 255) & ~(size_t)255;
        return o;
    };
    float* stats   = (float*)(ws + take(512 * 4));   // bn0 sum/sq (192+192), ens sum/sq (64+64)
    float* bn0_sum = stats;
    float* bn0_sq  = stats + 192;
    float* ens_sum = stats + 384;
    float* ens_sq  = stats + 448;
    float* normA = (float*)(ws + take((size_t)KK * GG * VV * VV * 4));
    float* Scol  = (float*)(ws + take((size_t)KK * GG * VV * 4));
    float* xm    = (float*)(ws + take((size_t)NN * CIN * VV * 4));
    float* a0    = (float*)(ws + take(D3 * 4));
    float* b0    = (float*)(ws + take(D3 * 4));
    float* a2    = (float*)(ws + take(COUT * 4));
    float* b2c   = (float*)(ws + take(COUT * 4));
    float* offb  = (float*)(ws + take((size_t)COUT * 32 * 4));
    _Float16* BscT  = (_Float16*)(ws + take((size_t)KK * COUT * 1024 * 2));
    _Float16* attT  = (_Float16*)(ws + take((size_t)KK * NN * COUT * 1024 * 2));
    _Float16* XlinP = (_Float16*)(ws + take((size_t)NN * D3 * TT * 32 * 2));
    _Float16* x3P   = (_Float16*)(ws + take((size_t)KK * NN * COUT * TT * 32 * 2));
    _Float16* catT  = (_Float16*)(ws + take((size_t)NN * TV * 128 * 2));
    _Float16* outp  = (_Float16*)(ws + take((size_t)NN * COUT * TV * 2));

    const float invM = 1.f / (float)MTOT;
    const long rowsX = (long)NN * D3 * TT;       // 3,145,728
    const long rows3 = (long)KK * NN * COUT * TT;

    k_zero<<<2, 256, 0, stream>>>(stats, 512);
    k_norm_adj<<<KK * GG, 32, 0, stream>>>(Dec, normA, Scol);
    k_mean<<<(NN * CIN * VV + 255) / 256, 256, 0, stream>>>(x0, xm);
    k_att<<<KK * NN, 256, 0, stream>>>(xm, w1, b1, w2, b2, w4, b4, alpha, Aadj, attT);
    k_padzero<<<(int)((rowsX + 255) / 256), 256, 0, stream>>>(XlinP, rowsX);
    k_padzero<<<(int)((rows3 + 255) / 256), 256, 0, stream>>>(x3P, rows3);

    k_lin_gemm<<<dim3(25, 12, NN), 256, 0, stream>>>(x0, Lw, Lb, XlinP, bn0_sum, bn0_sq);
    k_finalize_bn<<<1, 192, 0, stream>>>(bn0_sum, bn0_sq, g0, be0, a0, b0, D3, invM);
    k_bscale<<<KK * COUT, 256, 0, stream>>>(normA, a0, BscT);
    k_off<<<COUT, 32, 0, stream>>>(Scol, b0, offb);
    k_adj<<<dim3(TT / 16, COUT, NN), 64, 0, stream>>>(XlinP, BscT, offb, catT);

    k_x3_gemm<<<dim3(25, 12, NN), 256, 0, stream>>>(x0, w3, b3, x3P);
    k_y_gemm<<<dim3(TT / 16, COUT, NN), 64, 0, stream>>>(x3P, attT, catT);

    k_ens_gemm<<<dim3(25, 4, NN), 256, 0, stream>>>(catT, ew, eb, outp, ens_sum, ens_sq);
    k_finalize_bn<<<1, 64, 0, stream>>>(ens_sum, ens_sq, g1, be1, a2, b2c, COUT, invM);
    k_final<<<(int)(((size_t)NN * COUT * TV + 255) / 256), 256, 0, stream>>>(outp, x0, a2, b2c, out);
}
